// MultiHeadAttention_24893630447929
// MI455X (gfx1250) — matrix operand order repaired and checked by reading, not compiled, not run on hardware
//
#include <hip/hip_runtime.h>
#include <hip/hip_bf16.h>

// MHA forward for B=2,S=2048,D=1024,H=16,HD=64 on gfx1250 (wave32, WMMA).
// Pipeline: f32->bf16 convert -> QKV WMMA GEMMs (V stored transposed per head)
// -> RoPE -> flash attention (bf16 WMMA, fp32 accum, streaming softmax)
// -> O-proj WMMA GEMM (fp32 out). All tiles staged into LDS by the Tensor
// Data Mover (tensor_load_to_lds, TENSORcnt-tracked, double-buffered), with
// D# pad fields producing bank-conflict-avoiding LDS strides in hardware.

typedef __attribute__((ext_vector_type(16))) __bf16 v16bf;
typedef __attribute__((ext_vector_type(8)))  __bf16 v8bf;
typedef __attribute__((ext_vector_type(8)))  float  v8f;
typedef __attribute__((ext_vector_type(4)))  unsigned int u32x4;
typedef __attribute__((ext_vector_type(8)))  int i32x8;
typedef __attribute__((ext_vector_type(4)))  int i32x4;

#define SHUF16(lo, hi) __builtin_shufflevector((lo), (hi), 0,1,2,3,4,5,6,7,8,9,10,11,12,13,14,15)
#define WMMA_BF16(a, b, c) __builtin_amdgcn_wmma_f32_16x16x32_bf16(false, (a), false, (b), (short)0, (c), false, false)

constexpr int Bc = 2, Sq = 2048, Dm = 1024, Hn = 16, HD = 64;
constexpr int GM = Bc * Sq;   // 4096
constexpr int GN = Dm;        // 1024
constexpr int GK = Dm;        // 1024
constexpr int BM = 128, BN = 128, BK = 32;
constexpr int LSTR = 40;      // 32-elem rows: 16 dwords data + 4 dwords pad (80B)
constexpr int KSTR = 72;      // 64-elem rows: 32 dwords data + 4 dwords pad (144B)

__device__ __forceinline__ __bf16 f2bf(float f) {
  unsigned u = __builtin_bit_cast(unsigned, f);
  u += 0x7FFFu + ((u >> 16) & 1u);          // round-to-nearest-even
  unsigned short h = (unsigned short)(u >> 16);
  return __builtin_bit_cast(__bf16, h);
}

__device__ __forceinline__ float bf2f(__bf16 b) {
  unsigned short h = __builtin_bit_cast(unsigned short, b);
  unsigned u = (unsigned)h << 16;
  return __builtin_bit_cast(float, u);
}

__device__ __forceinline__ unsigned lds_off(const void* p) {
  // LDS aperture generic addresses carry the LDS byte offset in addr[31:0].
  return (unsigned)(unsigned long long)(uintptr_t)p;
}

// --------------------------------------------------------------- TDM load ---
// 2D tile load Global->LDS via Tensor Data Mover (D# per CDNA5 ISA).
__device__ __forceinline__ void tdm_load_2d(unsigned lds_addr, const void* gaddr,
                                            unsigned td0, unsigned td1,
                                            unsigned stride0,
                                            unsigned tile0, unsigned tile1,
                                            unsigned pad_int, unsigned pad_amt) {
  unsigned long long ga = (unsigned long long)(uintptr_t)gaddr;
  u32x4 g0;
  g0[0] = 1u;                                             // count=1, user D#
  g0[1] = lds_addr;                                       // lds_addr[31:0]
  g0[2] = (unsigned)(ga & 0xFFFFFFFFu);                   // global_addr[31:0]
  g0[3] = (unsigned)((ga >> 32) & 0x01FFFFFFu) | (2u << 30);  // addr[56:32], type=2
  i32x8 g1;
  g1[0] = (int)((1u << 16)                                // data_size = 2 bytes
              | (1u << 20)                                // pad_enable
              | (pad_int << 22) | (pad_amt << 25));
  g1[1] = (int)((td0 & 0xFFFFu) << 16);                   // tensor_dim0[15:0]
  g1[2] = (int)(((td0 >> 16) & 0xFFFFu) | ((td1 & 0xFFFFu) << 16));
  g1[3] = (int)(((td1 >> 16) & 0xFFFFu) | ((tile0 & 0xFFFFu) << 16));
  g1[4] = (int)(tile1 & 0xFFFFu);                         // tile_dim1; tile_dim2=0
  g1[5] = (int)stride0;                                   // dim0_stride[31:0]
  g1[6] = 0;                                              // dim0_stride[47:32]=0
  g1[7] = 0;                                              // dim1_stride unused (2D)
  i32x4 z4 = {0, 0, 0, 0};
#if __clang_major__ >= 23
  i32x8 z8 = {0, 0, 0, 0, 0, 0, 0, 0};
  __builtin_amdgcn_tensor_load_to_lds(g0, g1, z4, z4, z8, 0);
#else
  __builtin_amdgcn_tensor_load_to_lds(g0, g1, z4, z4, 0);
#endif
}

// ---------------------------------------------------------------- convert ---
__global__ void f32_to_bf16_kernel(const float* __restrict__ src,
                                   __bf16* __restrict__ dst, int n) {
  int i = blockIdx.x * blockDim.x + threadIdx.x;
  int stride = gridDim.x * blockDim.x;
  for (; i < n; i += stride) dst[i] = f2bf(src[i]);
}

// ------------------------------------------------------------------- GEMM ---
// C[GM x GN] = A[GM x GK](bf16,row) @ W[GN x GK](bf16,row)^T + bias
// out_mode 0: bf16 -> (B,H,S,HD)      (Q/K path)
// out_mode 1: fp32 -> row-major GMxGN (final output)
// out_mode 2: bf16 -> (B,H,HD,S)      (V^T path: contiguous keys per hd row)
__global__ __launch_bounds__(256) void gemm_bf16_kernel(
    const __bf16* __restrict__ A, const __bf16* __restrict__ W,
    const float* __restrict__ bias, void* __restrict__ out, int out_mode) {
  __shared__ __bf16 As[2][BM * LSTR];   // TDM-padded: 16 dwords data + 4 pad
  __shared__ __bf16 Ws[2][BN * LSTR];

  const int tid  = threadIdx.x;
  const int lane = tid & 31, wid = tid >> 5;
  const int half = lane >> 4, l16 = lane & 15;
  const int bm = blockIdx.y * BM, bn = blockIdx.x * BN;
  const int wm = (wid & 1) * 64, wn = (wid >> 1) * 32;   // 2x4 wave grid

  v8f acc[4][2];
#pragma unroll
  for (int mt = 0; mt < 4; ++mt)
#pragma unroll
    for (int nt = 0; nt < 2; ++nt) acc[mt][nt] = (v8f){0.f,0.f,0.f,0.f,0.f,0.f,0.f,0.f};

  const __bf16* Abase = A + (size_t)bm * GK;
  const __bf16* Wbase = W + (size_t)bn * GK;
  constexpr int nIter = GK / BK;   // 32

  // prologue: DMA first K-tile pair into buffer 0 (wave 0 issues, EXEC-free op)
  if (wid == 0) {
    tdm_load_2d(lds_off(&As[0][0]), Abase, GK, GM, GK, BK, BM, 3u, 3u);
    tdm_load_2d(lds_off(&Ws[0][0]), Wbase, GK, GN, GK, BK, BN, 3u, 3u);
  }

  for (int it = 0; it < nIter; ++it) {
    __syncthreads();   // compute(it-1) done -> buffer (it+1)&1 is reusable
    if (wid == 0) {
      if (it + 1 < nIter) {   // prefetch next tile pair via TDM, overlap compute
        tdm_load_2d(lds_off(&As[(it + 1) & 1][0]), Abase + (it + 1) * BK,
                    GK, GM, GK, BK, BM, 3u, 3u);
        tdm_load_2d(lds_off(&Ws[(it + 1) & 1][0]), Wbase + (it + 1) * BK,
                    GK, GN, GK, BK, BN, 3u, 3u);
        __builtin_amdgcn_s_wait_tensorcnt(2);  // 2 newest in flight => tile it done
      } else {
        __builtin_amdgcn_s_wait_tensorcnt(0);
      }
    }
    __syncthreads();   // tile it visible to all waves
    const __bf16* Ab = &As[it & 1][0];
    const __bf16* Wb = &Ws[it & 1][0];

    // A fragment: lane holds row (lane&15), K = kb+0..7 and kb+16..23, kb=8*half
    v16bf va[4], vb[2];
#pragma unroll
    for (int mt = 0; mt < 4; ++mt) {
      const __bf16* ap = &Ab[(wm + mt * 16 + l16) * LSTR + half * 8];
      v8bf lo = *(const v8bf*)ap;
      v8bf hi = *(const v8bf*)(ap + 16);
      va[mt] = SHUF16(lo, hi);
    }
    // B fragment: lane holds W row (col n = lane&15), K = kb+0..7 and kb+16..23, kb=8*half
#pragma unroll
    for (int nt = 0; nt < 2; ++nt) {
      const __bf16* wp = &Wb[(wn + nt * 16 + l16) * LSTR + half * 8];
      v8bf lo = *(const v8bf*)wp;
      v8bf hi = *(const v8bf*)(wp + 16);
      vb[nt] = SHUF16(lo, hi);
    }
#pragma unroll
    for (int mt = 0; mt < 4; ++mt)
#pragma unroll
      for (int nt = 0; nt < 2; ++nt)
        acc[mt][nt] = WMMA_BF16(va[mt], vb[nt], acc[mt][nt]);
  }

  // epilogue: C layout row = v + 8*half, col = lane&15
#pragma unroll
  for (int mt = 0; mt < 4; ++mt)
#pragma unroll
    for (int nt = 0; nt < 2; ++nt) {
      const int col = bn + wn + nt * 16 + l16;
      const float bv = bias[col];
#pragma unroll
      for (int v = 0; v < 8; ++v) {
        const int row = bm + wm + mt * 16 + v + 8 * half;
        const float val = acc[mt][nt][v] + bv;
        if (out_mode == 1) {
          ((float*)out)[(size_t)row * GN + col] = val;
        } else {
          const int b = row >> 11, s = row & (Sq - 1);
          const int h = col >> 6, dd = col & (HD - 1);
          if (out_mode == 0) {   // (B,H,S,HD)
            ((__bf16*)out)[((((size_t)b * Hn + h) * Sq + s) << 6) + dd] = f2bf(val);
          } else {               // (B,H,HD,S) : transposed V
            ((__bf16*)out)[((((size_t)b * Hn + h) * HD + dd) << 11) + s] = f2bf(val);
          }
        }
      }
    }
}

// ------------------------------------------------------------------- RoPE ---
// In-place RoPE on bf16 tensor (B,H,S,HD). One thread per (bh,s,dd<32) pair.
__global__ void rope_kernel(__bf16* __restrict__ t) {
  const int idx = blockIdx.x * blockDim.x + threadIdx.x;  // B*H*S*32 threads
  const int dd = idx & 31;
  const int s  = (idx >> 5) & (Sq - 1);
  const int bh = idx >> 16;
  __bf16* p = t + (size_t)bh * Sq * HD + (size_t)s * HD;
  const float x1 = bf2f(p[dd]);
  const float x2 = bf2f(p[dd + 32]);
  // inv_freq = 10000^{-2*dd/64} = exp(-dd/32 * ln(1e4))
  const float inv_freq = __expf(-(float)dd * (9.210340371976184f / 32.0f));
  const float th = (float)s * inv_freq;
  const float c = __cosf(th), sn = __sinf(th);
  p[dd]      = f2bf(x1 * c - x2 * sn);
  p[dd + 32] = f2bf(x2 * c + x1 * sn);
}

// -------------------------------------------------------- flash attention ---
// Block = (64 q-rows, h, b); 4 waves x 16 q-rows. Streaming softmax over S
// in 32-key steps; scores & PV via bf16 WMMA with fp32 accumulators.
// K (S x HD) and V^T (HD x S) tiles DMA'd by TDM, double-buffered.
__global__ __launch_bounds__(128) void attn_kernel(
    const __bf16* __restrict__ Q, const __bf16* __restrict__ Kt,
    const __bf16* __restrict__ VT, __bf16* __restrict__ ctx) {
  __shared__ __bf16 Ks[2][32 * KSTR];   // 32 keys x 64 hd, stride 72
  __shared__ __bf16 Vs[2][HD * LSTR];   // 64 hd rows x 32 keys, stride 40
  __shared__ __bf16 Ps[4 * 16 * 40];    // per-wave 16x32 P tile, stride 40

  const int tid = threadIdx.x, lane = tid & 31, wid = tid >> 5;
  const int half = lane >> 4, l16 = lane & 15;
  const int qt = blockIdx.x, h = blockIdx.y, b = blockIdx.z;
  const size_t base = ((size_t)b * Hn + h) * Sq * HD;   // same extent for K and VT
  const int qrow0 = qt * 64 + wid * 16;

  // Q fragments (16 rows x 64 d) -> 2 A-frags, K-dim 32 each
  v16bf qa[2];
  {
    const __bf16* qp = Q + base + (size_t)(qrow0 + l16) * HD;
#pragma unroll
    for (int d = 0; d < 2; ++d) {
      const __bf16* p = qp + d * 32 + half * 8;
      v8bf lo = *(const v8bf*)p;
      v8bf hi = *(const v8bf*)(p + 16);
      qa[d] = SHUF16(lo, hi);
    }
  }

  float mi[8], li[8];
  v8f acc[4];
#pragma unroll
  for (int v = 0; v < 8; ++v) { mi[v] = -1e30f; li[v] = 0.f; }
#pragma unroll
  for (int nt = 0; nt < 4; ++nt) acc[nt] = (v8f){0.f,0.f,0.f,0.f,0.f,0.f,0.f,0.f};

  const __bf16* Kbase = Kt + base;   // (S x HD) row-major
  const __bf16* Vbase = VT + base;   // (HD x S) row-major
  constexpr int nIter = Sq / 32;     // 64

  if (wid == 0) {
    tdm_load_2d(lds_off(&Ks[0][0]), Kbase, HD, Sq, HD, HD, 32, 4u, 3u);
    tdm_load_2d(lds_off(&Vs[0][0]), Vbase, Sq, HD, Sq, 32, HD, 3u, 3u);
  }

  for (int it = 0; it < nIter; ++it) {
    __syncthreads();   // compute(it-1) done -> buffer (it+1)&1 reusable
    if (wid == 0) {
      if (it + 1 < nIter) {
        tdm_load_2d(lds_off(&Ks[(it + 1) & 1][0]), Kbase + (size_t)(it + 1) * 32 * HD,
                    HD, Sq, HD, HD, 32, 4u, 3u);
        tdm_load_2d(lds_off(&Vs[(it + 1) & 1][0]), Vbase + (it + 1) * 32,
                    Sq, HD, Sq, 32, HD, 3u, 3u);
        __builtin_amdgcn_s_wait_tensorcnt(2);
      } else {
        __builtin_amdgcn_s_wait_tensorcnt(0);
      }
    }
    __syncthreads();   // K/V tile it ready
    const __bf16* Kb = &Ks[it & 1][0];
    const __bf16* Vb = &Vs[it & 1][0];

    // scores: two 16x16 tiles (keys 32*it..+15, +16..+31)
    v8f sc[2];
    sc[0] = (v8f){0.f,0.f,0.f,0.f,0.f,0.f,0.f,0.f};
    sc[1] = sc[0];
#pragma unroll
    for (int nt = 0; nt < 2; ++nt)
#pragma unroll
      for (int d = 0; d < 2; ++d) {
        const __bf16* kp = &Kb[(nt * 16 + l16) * KSTR + d * 32 + half * 8];
        v8bf lo = *(const v8bf*)kp;
        v8bf hi = *(const v8bf*)(kp + 16);
        v16bf kf = SHUF16(lo, hi);
        sc[nt] = WMMA_BF16(qa[d], kf, sc[nt]);
      }

    // online softmax; row = v + 8*half, cols across the 16-lane group
    float fct[8], p0[8], p1[8];
#pragma unroll
    for (int v = 0; v < 8; ++v) {
      float s0 = sc[0][v] * 0.125f;   // 1/sqrt(64)
      float s1 = sc[1][v] * 0.125f;
      float rm = fmaxf(s0, s1);
      rm = fmaxf(rm, __shfl_xor(rm, 1));
      rm = fmaxf(rm, __shfl_xor(rm, 2));
      rm = fmaxf(rm, __shfl_xor(rm, 4));
      rm = fmaxf(rm, __shfl_xor(rm, 8));
      const float mn = fmaxf(mi[v], rm);
      const float fc = __expf(mi[v] - mn);
      const float e0 = __expf(s0 - mn);
      const float e1 = __expf(s1 - mn);
      float rs = e0 + e1;
      rs += __shfl_xor(rs, 1);
      rs += __shfl_xor(rs, 2);
      rs += __shfl_xor(rs, 4);
      rs += __shfl_xor(rs, 8);
      li[v] = li[v] * fc + rs;
      mi[v] = mn;
      fct[v] = fc; p0[v] = e0; p1[v] = e1;
    }
#pragma unroll
    for (int nt = 0; nt < 4; ++nt)
#pragma unroll
      for (int v = 0; v < 8; ++v) acc[nt][v] *= fct[v];

    // C-layout -> A-layout transpose of P through wave-private LDS
    __bf16* Pw = &Ps[wid * 16 * 40];
#pragma unroll
    for (int v = 0; v < 8; ++v) {
      const int r = v + 8 * half;
      Pw[r * 40 + l16]      = f2bf(p0[v]);
      Pw[r * 40 + 16 + l16] = f2bf(p1[v]);
    }
    v16bf pa;
    {
      const __bf16* pp = &Pw[l16 * 40 + half * 8];
      v8bf lo = *(const v8bf*)pp;
      v8bf hi = *(const v8bf*)(pp + 16);
      pa = SHUF16(lo, hi);
    }

    // PV: O(16x64) += P(16x32) @ V(32x64); B column n = VT row n (contiguous)
#pragma unroll
    for (int nt = 0; nt < 4; ++nt) {
      const __bf16* vp = &Vb[(nt * 16 + l16) * LSTR + half * 8];
      v8bf lo = *(const v8bf*)vp;
      v8bf hi = *(const v8bf*)(vp + 16);
      v16bf bvv = SHUF16(lo, hi);
      acc[nt] = WMMA_BF16(pa, bvv, acc[nt]);
    }
  }

  // normalize and write ctx as (B,S,D) bf16 for the O projection
#pragma unroll
  for (int nt = 0; nt < 4; ++nt)
#pragma unroll
    for (int v = 0; v < 8; ++v) {
      const int r = v + 8 * half;
      const int s = qt * 64 + wid * 16 + r;
      const float val = acc[nt][v] / li[v];
      ctx[((size_t)b * Sq + s) * Dm + h * HD + nt * 16 + l16] = f2bf(val);
    }
}

// ----------------------------------------------------------------- launch ---
extern "C" void kernel_launch(void* const* d_in, const int* in_sizes, int n_in,
                              void* d_out, int out_size, void* d_ws, size_t ws_size,
                              hipStream_t stream) {
  (void)in_sizes; (void)n_in; (void)out_size; (void)ws_size;
  const float* x  = (const float*)d_in[0];
  const float* wq = (const float*)d_in[1];
  const float* bq = (const float*)d_in[2];
  const float* wk = (const float*)d_in[3];
  const float* bk = (const float*)d_in[4];
  const float* wv = (const float*)d_in[5];
  const float* bv = (const float*)d_in[6];
  const float* wo = (const float*)d_in[7];
  const float* bo = (const float*)d_in[8];
  float* out = (float*)d_out;

  char* ws = (char*)d_ws;
  const size_t MB = 1u << 20;
  __bf16* xb  = (__bf16*)(ws + 0 * MB);    // 8 MB  : x bf16 (4096x1024)
  __bf16* wqb = (__bf16*)(ws + 8 * MB);    // 2 MB
  __bf16* wkb = (__bf16*)(ws + 10 * MB);   // 2 MB
  __bf16* wvb = (__bf16*)(ws + 12 * MB);   // 2 MB
  __bf16* wob = (__bf16*)(ws + 14 * MB);   // 2 MB
  __bf16* qb  = (__bf16*)(ws + 16 * MB);   // 8 MB  : (B,H,S,HD)
  __bf16* kb  = (__bf16*)(ws + 24 * MB);   // 8 MB  : (B,H,S,HD)
  __bf16* vb  = (__bf16*)(ws + 32 * MB);   // 8 MB  : (B,H,HD,S)  V transposed
  __bf16* cb  = (__bf16*)(ws + 40 * MB);   // 8 MB  : ctx (B,S,D) -> 48 MB total

  f32_to_bf16_kernel<<<4096, 256, 0, stream>>>(x,  xb,  GM * GK);
  f32_to_bf16_kernel<<<4096, 256, 0, stream>>>(wq, wqb, GN * GK);
  f32_to_bf16_kernel<<<4096, 256, 0, stream>>>(wk, wkb, GN * GK);
  f32_to_bf16_kernel<<<4096, 256, 0, stream>>>(wv, wvb, GN * GK);
  f32_to_bf16_kernel<<<4096, 256, 0, stream>>>(wo, wob, GN * GK);

  dim3 gg(GN / BN, GM / BM);  // (8, 32)
  gemm_bf16_kernel<<<gg, 256, 0, stream>>>(xb, wqb, bq, qb, 0);
  gemm_bf16_kernel<<<gg, 256, 0, stream>>>(xb, wkb, bk, kb, 0);
  gemm_bf16_kernel<<<gg, 256, 0, stream>>>(xb, wvb, bv, vb, 2);

  const int rope_threads = Bc * Hn * Sq * 32;  // 2,097,152
  rope_kernel<<<rope_threads / 256, 256, 0, stream>>>(qb);
  rope_kernel<<<rope_threads / 256, 256, 0, stream>>>(kb);

  attn_kernel<<<dim3(Sq / 64, Hn, Bc), 128, 0, stream>>>(qb, kb, vb, cb);

  gemm_bf16_kernel<<<gg, 256, 0, stream>>>(cb, wob, bo, out, 1);
}
